// EnhancedMemoryUnit_2405181686368
// MI455X (gfx1250) — matrix operand order repaired and checked by reading, not compiled, not run on hardware
//
#include <hip/hip_runtime.h>
#include <hip/hip_bf16.h>
#include <math.h>

// ---------------------------------------------------------------------------
// Fused "EnhancedMemoryUnit" for MI455X (gfx1250, wave32, WMMA).
//   out[b,p,:] = softmax(X[b,:] . W[p,:,:]^T) . W[p,:,:]
// bf16 WMMA (v_wmma_f32_16x16x32_bf16) with f32 accumulation, fully fused.
// Both GEMM passes: 8-way accumulator blocking + explicit cross-kt double
// buffering of operands (prefetch distance = one full 8-WMMA burst), and
// single-base addressing (constant tile strides folded into the 24-bit
// instruction offset) to minimize address VALU / VGPR pressure.
// ---------------------------------------------------------------------------

typedef __bf16 bf16;
typedef __attribute__((ext_vector_type(16))) __bf16 v16bf;
typedef __attribute__((ext_vector_type(8)))  float  v8f;

constexpr int B = 8192, P = 8, M = 2048, C = 1024;
constexpr int BT = 16;                       // batch rows per block
constexpr size_t XBF_BYTES = (size_t)B * C * sizeof(bf16);
constexpr size_t WBF_BYTES = (size_t)P * M * C * sizeof(bf16);

// LDS carve-up for the main kernel (dynamic shared memory)
constexpr int LDS_XS   = BT * C * 2;         // 32 KB  bf16 X tile [16][1024]
constexpr int LDS_S    = BT * M * 4;         // 128 KB f32 scores  [16][2048]
constexpr int LDS_PB   = BT * M * 2;         // 64 KB  bf16 probs  [16][2048]
constexpr int LDS_INV  = BT * 4;             // 64 B   f32 1/rowsum
constexpr int SMEM_MAIN = LDS_XS + LDS_S + LDS_PB + LDS_INV;   // 229,440 B

union ABv { v16bf v; int4 q[2]; };           // 16 bf16 = two b128 chunks

// load 8 B-operand tiles; JS = constant byte stride between tiles
template <int JS>
__device__ __forceinline__ void load_b8(ABv bb[8], const char* base, int kt) {
    #pragma unroll
    for (int j = 0; j < 8; ++j) {
        bb[j].q[0] = *(const int4*)(base + j * JS + kt * 64);
        bb[j].q[1] = *(const int4*)(base + j * JS + kt * 64 + 16);
    }
}

__device__ __forceinline__ void load_a(ABv& a, const char* aptr, int kt) {
    a.q[0] = *(const int4*)(aptr + kt * 64);
    a.q[1] = *(const int4*)(aptr + kt * 64 + 16);
}

__device__ __forceinline__ void burst8(v8f acc[8], const ABv& a, const ABv bb[8]) {
    #pragma unroll
    for (int j = 0; j < 8; ++j)
        acc[j] = __builtin_amdgcn_wmma_f32_16x16x32_bf16(
            false, a.v, false, bb[j].v, (short)0, acc[j], false, false);
}

// Double-buffered K loop: while bursting WMMAs on one operand set, the other
// set's 16 b128 loads are already in flight.
template <int JS>
__device__ __forceinline__ void gemm_k_loop(v8f acc[8], const char* aptr,
                                            const char* bbase, int NK) {
    ABv bb0[8], bb1[8], a0, a1;
    load_b8<JS>(bb0, bbase, 0);
    load_a(a0, aptr, 0);
    int kt = 0;
    for (; kt < NK - 2; kt += 2) {
        load_b8<JS>(bb1, bbase, kt + 1);
        load_a(a1, aptr, kt + 1);
        burst8(acc, a0, bb0);
        load_b8<JS>(bb0, bbase, kt + 2);
        load_a(a0, aptr, kt + 2);
        burst8(acc, a1, bb1);
    }
    // tail: kt == NK-2
    load_b8<JS>(bb1, bbase, NK - 1);
    load_a(a1, aptr, NK - 1);
    burst8(acc, a0, bb0);
    burst8(acc, a1, bb1);
}

// -------------------------------- stage 0 ----------------------------------
__global__ void emu_convert_x(const float* __restrict__ X,
                              bf16* __restrict__ Xbf, int n4) {
    int i = blockIdx.x * blockDim.x + threadIdx.x;
    if (i >= n4) return;
    float4 v = ((const float4*)X)[i];
    union { bf16 b[4]; uint2 u; } o;
    o.b[0] = (bf16)v.x; o.b[1] = (bf16)v.y;
    o.b[2] = (bf16)v.z; o.b[3] = (bf16)v.w;
    ((uint2*)Xbf)[i] = o.u;
}

// f32 W[p][m][c] -> bf16 W[p][m][c]  and  bf16 Wt[p][c][m]  (LDS-tiled transpose)
__global__ void emu_convert_w(const float* __restrict__ W,
                              bf16* __restrict__ Wbf, bf16* __restrict__ Wt) {
    __shared__ bf16 tile[32][33];
    const int p  = blockIdx.z;
    const int m0 = blockIdx.y * 32;
    const int c0 = blockIdx.x * 32;
    const int tx = threadIdx.x;              // 0..31 (along c on read, m on write)
    const int ty = threadIdx.y;              // 0..7
    #pragma unroll
    for (int i = ty; i < 32; i += 8) {
        size_t idx = ((size_t)(p * M + m0 + i)) * C + c0 + tx;
        bf16 b = (bf16)W[idx];
        Wbf[idx] = b;
        tile[i][tx] = b;                     // tile[m_local][c_local]
    }
    __syncthreads();
    #pragma unroll
    for (int i = ty; i < 32; i += 8) {
        // Wt[p][c0+i][m0+tx] = W[p][m0+tx][c0+i]
        Wt[((size_t)(p * C + c0 + i)) * M + m0 + tx] = tile[tx][i];
    }
}

// ------------------------------- main kernel -------------------------------
__global__ __launch_bounds__(256)
void emu_main(const bf16* __restrict__ Xbf, const bf16* __restrict__ Wbf,
              const bf16* __restrict__ Wt, float* __restrict__ out) {
    extern __shared__ char smem[];
    bf16*  Xs   = (bf16*)smem;                                   // [16][C]
    float* S    = (float*)(smem + LDS_XS);                       // [16][M]
    bf16*  Pb   = (bf16*)(smem + LDS_XS + LDS_S);                // [16][M]
    float* invs = (float*)(smem + LDS_XS + LDS_S + LDS_PB);      // [16]

    const int bt   = blockIdx.x;             // batch tile
    const int p    = blockIdx.y;             // head
    const int tid  = threadIdx.x;
    const int wave = tid >> 5;               // 0..7
    const int lane = tid & 31;
    const int hi   = lane >> 4;              // lane-half (K split)
    const int ln   = lane & 15;              // N / row-within-half

    // ---- stage X tile into LDS (coalesced b128) ----
    {
        const int4* xg  = (const int4*)(Xbf + (size_t)bt * BT * C);
        int4*       xs4 = (int4*)Xs;
        #pragma unroll
        for (int i = 0; i < (BT * C * 2 / 16) / 256; ++i)        // 8 iters
            xs4[tid + i * 256] = xg[tid + i * 256];
    }
    __syncthreads();

    // ---- pass 1: S[16 x M] = X . W^T ----
    // Wave w owns m in [w*256, w*256+256): 16 m-tiles as 2 groups of 8.
    // B tile j has constant stride 16*C*2 = 32 KB (fits 24-bit inst offset).
    for (int t = 0; t < 2; ++t) {
        v8f acc[8];
        {
            v8f z = {};
            #pragma unroll
            for (int j = 0; j < 8; ++j) acc[j] = z;
        }
        const char* aptr  = (const char*)Xs + ((ln * C + hi * 16) * 2);
        const int   mt0   = wave * 16 + t * 8;                   // first m-tile
        const char* bbase = (const char*)Wbf +
            (((size_t)(p * M + mt0 * 16 + ln)) * C + hi * 16) * 2;
        gemm_k_loop<16 * C * 2>(acc, aptr, bbase, C / 32);
        #pragma unroll
        for (int j = 0; j < 8; ++j) {
            const int mt = mt0 + j;
            #pragma unroll
            for (int r = 0; r < 8; ++r)      // D: row b=r+8*hi, col m=ln
                S[(r + 8 * hi) * M + mt * 16 + ln] = acc[j][r];
        }
    }
    __syncthreads();

    // ---- softmax over M per row; store UNNORMALIZED exp as bf16 ----
    {
        const int row = tid >> 4;            // 0..15
        const int sub = tid & 15;
        float mx = -INFINITY;
        for (int j = sub; j < M; j += 16)
            mx = fmaxf(mx, S[row * M + j]);
        #pragma unroll
        for (int off = 8; off >= 1; off >>= 1)
            mx = fmaxf(mx, __shfl_xor(mx, off, 32));   // 16-lane group reduce
        float sum = 0.f;
        for (int j = sub; j < M; j += 16) {
            float e = __expf(S[row * M + j] - mx);
            sum += e;
            Pb[row * M + j] = (bf16)e;
        }
        #pragma unroll
        for (int off = 8; off >= 1; off >>= 1)
            sum += __shfl_xor(sum, off, 32);
        if (sub == 0) invs[row] = 1.0f / sum;
    }
    __syncthreads();

    // ---- pass 2: O[16 x C] = P . W ; each wave owns 8 column tiles ----
    // B tile j has constant stride 16*M*2 = 64 KB (fits 24-bit inst offset).
    v8f acc2[8];
    {
        v8f z = {};
        #pragma unroll
        for (int j = 0; j < 8; ++j) acc2[j] = z;
    }
    {
        const char* aptr2  = (const char*)Pb + ((ln * M + hi * 16) * 2);
        const int   n0     = wave * 8 * 16 + ln;       // first output column
        const char* bbase2 = (const char*)Wt +
            (((size_t)(p * C + n0)) * M + hi * 16) * 2;
        gemm_k_loop<16 * M * 2>(acc2, aptr2, bbase2, M / 32);
    }

    // ---- epilogue: fold 1/rowsum, coalesced f32 stores to out[B][P][C] ----
    float iv[8];
    #pragma unroll
    for (int r = 0; r < 8; ++r) iv[r] = invs[r + 8 * hi];
    #pragma unroll
    for (int j = 0; j < 8; ++j) {
        const int n0 = (wave * 8 + j) * 16 + ln;
        #pragma unroll
        for (int r = 0; r < 8; ++r) {
            const int brow = bt * BT + r + 8 * hi;
            out[((size_t)brow * P + p) * C + n0] = acc2[j][r] * iv[r];
        }
    }
}

// ------------------------------- launcher ----------------------------------
extern "C" void kernel_launch(void* const* d_in, const int* in_sizes, int n_in,
                              void* d_out, int out_size, void* d_ws, size_t ws_size,
                              hipStream_t stream) {
    const float* X = (const float*)d_in[0];   // [B, C]  f32
    const float* W = (const float*)d_in[1];   // [P, M, C] f32
    float* out = (float*)d_out;               // [B, P, C] f32
    (void)in_sizes; (void)n_in; (void)out_size; (void)ws_size;

    char* ws  = (char*)d_ws;
    bf16* Xbf = (bf16*)ws;                               // B*C bf16
    bf16* Wbf = (bf16*)(ws + XBF_BYTES);                 // P*M*C bf16 [m][c]
    bf16* Wt  = (bf16*)(ws + XBF_BYTES + WBF_BYTES);     // P*C*M bf16 [c][m]

    // one-time (per call) bf16 conversion + weight transpose
    {
        const int n4 = B * C / 4;
        emu_convert_x<<<(n4 + 255) / 256, 256, 0, stream>>>(X, Xbf, n4);
        dim3 g(C / 32, M / 32, P), blk(32, 8);
        emu_convert_w<<<g, blk, 0, stream>>>(W, Wbf, Wt);
    }

    // fused scores -> softmax -> combine
    (void)hipFuncSetAttribute(reinterpret_cast<const void*>(emu_main),
                              hipFuncAttributeMaxDynamicSharedMemorySize,
                              SMEM_MAIN);
    emu_main<<<dim3(B / BT, P), 256, SMEM_MAIN, stream>>>(Xbf, Wbf, Wt, out);
}
